// RetrievalModule_38963943309327
// MI455X (gfx1250) — matrix operand order repaired and checked by reading, not compiled, not run on hardware
//
#include <hip/hip_runtime.h>

// ---------------- types ----------------
typedef __attribute__((ext_vector_type(16))) _Float16 v16h;
typedef __attribute__((ext_vector_type(8)))  _Float16 v8h;
typedef __attribute__((ext_vector_type(8)))  float    v8f;

union Frag { v16h f16; v8h h[2]; };

#define NEGV   (-1e30f)
#define VALIDT (-5e29f)
#define EPSN   (1e-8f)

__device__ __forceinline__ v8f wmma_f16(v16h a, v16h b, v8f c) {
  return __builtin_amdgcn_wmma_f32_16x16x32_f16(false, a, false, b, (short)0, c, false, false);
}

__device__ __forceinline__ void top5_insert(float v, int idx, float* tv, int* ti) {
  if (v > tv[4]) {
    int j = 4;
    if (v > tv[3]) { tv[4]=tv[3]; ti[4]=ti[3]; j=3;
      if (v > tv[2]) { tv[3]=tv[2]; ti[3]=ti[2]; j=2;
        if (v > tv[1]) { tv[2]=tv[1]; ti[2]=ti[1]; j=1;
          if (v > tv[0]) { tv[1]=tv[0]; ti[1]=ti[0]; j=0; } } } }
    tv[j]=v; ti[j]=idx;
  }
}

// ---------------- kernel 1: normalize queries -> f16 ----------------
__global__ __launch_bounds__(256)
void qnorm_kernel(const float* __restrict__ q, _Float16* __restrict__ qh) {
  const int row = blockIdx.x, tid = threadIdx.x;
  __shared__ float red[256];
  const float* src = q + (size_t)row * 768;
  float s = 0.f;
  #pragma unroll
  for (int t = 0; t < 3; ++t) { float x = src[tid + t*256]; s += x*x; }
  red[tid] = s; __syncthreads();
  for (int o = 128; o > 0; o >>= 1) {
    if (tid < o) red[tid] += red[tid + o];
    __syncthreads();
  }
  const float inv = 1.f / fmaxf(sqrtf(red[0]), EPSN);
  _Float16* dst = qh + (size_t)row * 768;
  #pragma unroll
  for (int t = 0; t < 3; ++t) { int d = tid + t*256; dst[d] = (_Float16)(src[d] * inv); }
}

// ---------------- kernel 2: f32 -> f16 elementwise (weights) ----------------
__global__ __launch_bounds__(256)
void cvt_f16_kernel(const float* __restrict__ in, _Float16* __restrict__ out, int n) {
  int i = blockIdx.x * 256 + threadIdx.x;
  if (i < n) out[i] = (_Float16)in[i];
}

// ---------------- kernel 3: similarity (WMMA) + mask + per-chunk top-5 ----------------
// Block: 512 threads (16 waves), 64 training columns per block.
// LDS: sB 96KB (B-fragments, reused as C scratch), + spk/inv/sum (64 each).
__global__ __launch_bounds__(512)
void sim_topk_kernel(const _Float16* __restrict__ qnh,
                     const float* __restrict__ train,
                     const int* __restrict__ spk,
                     const int* __restrict__ tgt,
                     float* __restrict__ candV, int* __restrict__ candI,
                     int N, int nChunk)
{
  extern __shared__ char smem[];
  char*  sB   = smem;                          // 98304 bytes
  int*   sSpk = (int*)  (smem + 98304);        // 64 ints
  float* sInv = (float*)(smem + 98304 + 256);  // 64 floats
  float* sSum = (float*)(smem + 98304 + 512);  // 64 floats
  float* scratch = (float*)smem;               // reuse after barrier (64KB used)

  const int tid  = threadIdx.x;
  const int lane = tid & 31;
  const int w    = tid >> 5;          // wave id 0..15 -> row tile
  const int chunk = blockIdx.x;
  const int c0    = chunk * 64;

  if (tid < 64) {
    sSum[tid] = 0.f;
    int g = c0 + tid;
    sSpk[tid] = (g < N) ? spk[g] : -1;
  }
  __syncthreads();

  // ---- Phase A: load f32 features once, convert to f16 B-fragment layout, fuse sumsq ----
  {
    const int col  = tid >> 3;        // 0..63
    const int part = tid & 7;         // 0..7 (each covers 96 k)
    const int gcol = c0 + col;
    const int sub  = col >> 4, cw = col & 15;
    const float* src = train + (size_t)gcol * 768;
    float acc = 0.f;
    #pragma unroll
    for (int j = 0; j < 6; ++j) {
      const int k0 = part * 96 + j * 16;
      float4 ff[4];
      if (gcol < N) {
        ff[0] = *(const float4*)(src + k0);
        ff[1] = *(const float4*)(src + k0 + 4);
        ff[2] = *(const float4*)(src + k0 + 8);
        ff[3] = *(const float4*)(src + k0 + 12);
      } else {
        ff[0] = ff[1] = ff[2] = ff[3] = make_float4(0.f, 0.f, 0.f, 0.f);
      }
      const float* f = (const float*)ff;
      v8h h0, h1;
      #pragma unroll
      for (int i = 0; i < 8; ++i) { h0[i] = (_Float16)f[i]; h1[i] = (_Float16)f[8 + i]; }
      #pragma unroll
      for (int i = 0; i < 16; ++i) acc += f[i] * f[i];
      // B layout: lanes 0-15 hold K = 32t..32t+15 (col = lane), lanes 16-31 hold +16
      const int kstep = k0 >> 5;
      const int upper = (k0 >> 4) & 1;
      const int bl    = cw + upper * 16;
      v8h* dst = (v8h*)(sB + (((sub * 24 + kstep) * 32 + bl) << 5));
      dst[0] = h0; dst[1] = h1;
    }
    atomicAdd(&sSum[col], acc);
  }
  __syncthreads();
  if (tid < 64) sInv[tid] = 1.f / fmaxf(sqrtf(sSum[tid]), EPSN);
  __syncthreads();

  // ---- Phase B: WMMA. wave w -> rows [16w, 16w+16), 4 column subtiles ----
  v8f C[4] = {};
  {
    const _Float16* ap = qnh + (size_t)(w * 16 + (lane & 15)) * 768 + ((lane >> 4) << 4);
    for (int ks = 0; ks < 24; ++ks) {
      Frag A;
      A.h[0] = *(const v8h*)(ap + ks * 32);
      A.h[1] = *(const v8h*)(ap + ks * 32 + 8);
      #pragma unroll
      for (int s = 0; s < 4; ++s) {
        Frag B;
        const v8h* bp = (const v8h*)(sB + (((s * 24 + ks) * 32 + lane) << 5));
        B.h[0] = bp[0]; B.h[1] = bp[1];
        C[s] = wmma_f16(A.f16, B.f16, C[s]);
      }
    }
  }
  __syncthreads();   // everyone done reading sB

  // ---- Phase C: dump C to LDS scratch ----
  #pragma unroll
  for (int s = 0; s < 4; ++s) {
    #pragma unroll
    for (int v = 0; v < 8; ++v) {
      const int m   = v + ((lane >> 4) << 3);
      const int col = s * 16 + (lane & 15);
      scratch[(w << 10) + m * 64 + col] = C[s][v];
    }
  }
  __syncthreads();

  // ---- scan: lane l<16 -> row m=l cols[0,32); lane>=16 -> row m=l-16 cols[32,64) ----
  {
    const int m  = lane & 15;
    const int hf = lane >> 4;
    const int r  = w * 16 + m;
    const int tsid = tgt[r];
    float tv[5]; int ti[5];
    #pragma unroll
    for (int j = 0; j < 5; ++j) { tv[j] = NEGV; ti[j] = -1; }
    for (int c = hf * 32; c < hf * 32 + 32; ++c) {
      const int gcol = c0 + c;
      if (gcol < N && sSpk[c] == tsid) {
        float v = scratch[(w << 10) + m * 64 + c] * sInv[c];
        top5_insert(v, gcol, tv, ti);
      }
    }
    // merge halves: lane r pulls lane r+16's list
    #pragma unroll
    for (int j = 0; j < 5; ++j) {
      float ov = __shfl(tv[j], lane + 16);
      int   oi = __shfl(ti[j], lane + 16);
      if (lane < 16) top5_insert(ov, oi, tv, ti);
    }
    if (lane < 16) {
      size_t base = ((size_t)r * nChunk + chunk) * 5;
      #pragma unroll
      for (int j = 0; j < 5; ++j) { candV[base + j] = tv[j]; candI[base + j] = ti[j]; }
    }
  }
}

// ---------------- kernel 4: merge candidates, gather, build f16 combined ----------------
__global__ __launch_bounds__(256)
void merge_gather_kernel(const float* __restrict__ candV, const int* __restrict__ candI,
                         const float* __restrict__ train, const float* __restrict__ content,
                         _Float16* __restrict__ comb, int* __restrict__ hasMatch,
                         int nChunk)
{
  __shared__ float sV[256 * 5]; __shared__ int sI[256 * 5];
  __shared__ float mV[32 * 5];  __shared__ int mI[32 * 5];
  __shared__ float fV[5];       __shared__ int fI[5];
  __shared__ int sCnt;

  const int row = blockIdx.x, tid = threadIdx.x;
  const int nCand = nChunk * 5;
  const size_t base = (size_t)row * nCand;

  float tv[5]; int ti[5];
  #pragma unroll
  for (int j = 0; j < 5; ++j) { tv[j] = NEGV; ti[j] = -1; }
  for (int i = tid; i < nCand; i += 256) top5_insert(candV[base + i], candI[base + i], tv, ti);
  #pragma unroll
  for (int j = 0; j < 5; ++j) { sV[tid * 5 + j] = tv[j]; sI[tid * 5 + j] = ti[j]; }
  __syncthreads();

  if (tid < 32) {
    #pragma unroll
    for (int j = 0; j < 5; ++j) { tv[j] = NEGV; ti[j] = -1; }
    for (int q = tid * 8; q < tid * 8 + 8; ++q)
      #pragma unroll
      for (int j = 0; j < 5; ++j) top5_insert(sV[q * 5 + j], sI[q * 5 + j], tv, ti);
    #pragma unroll
    for (int j = 0; j < 5; ++j) { mV[tid * 5 + j] = tv[j]; mI[tid * 5 + j] = ti[j]; }
  }
  __syncthreads();

  if (tid == 0) {
    #pragma unroll
    for (int j = 0; j < 5; ++j) { tv[j] = NEGV; ti[j] = -1; }
    for (int q = 0; q < 32; ++q)
      #pragma unroll
      for (int j = 0; j < 5; ++j) top5_insert(mV[q * 5 + j], mI[q * 5 + j], tv, ti);
    int cnt = 0;
    #pragma unroll
    for (int j = 0; j < 5; ++j) {
      fV[j] = tv[j]; fI[j] = ti[j];
      if (tv[j] > VALIDT && ti[j] >= 0) ++cnt;
    }
    sCnt = cnt;
    hasMatch[row] = (cnt > 0) ? 1 : 0;
  }
  __syncthreads();

  const float denom = (float)max(sCnt, 1);
  const float* crow = content + (size_t)row * 768;
  _Float16* drow = comb + (size_t)row * 1536;
  #pragma unroll
  for (int t = 0; t < 3; ++t) {
    const int d = tid + t * 256;
    float s = 0.f;
    #pragma unroll
    for (int j = 0; j < 5; ++j)
      if (fV[j] > VALIDT && fI[j] >= 0) s += train[(size_t)fI[j] * 768 + d];
    drow[d]       = (_Float16)crow[d];
    drow[768 + d] = (_Float16)(s / denom);
  }
}

// ---------------- kernel 5: h = relu(combined @ W1^T + b1) (WMMA, K=1536) ----------------
__global__ __launch_bounds__(32)
void mlp1_kernel(const _Float16* __restrict__ comb, const _Float16* __restrict__ w1h,
                 const float* __restrict__ b1, _Float16* __restrict__ hh)
{
  const int lane = threadIdx.x;
  const int nt = blockIdx.x, rt = blockIdx.y;
  const _Float16* ap = comb + (size_t)(rt * 16 + (lane & 15)) * 1536 + ((lane >> 4) << 3);
  const _Float16* bp = w1h  + (size_t)(nt * 16 + (lane & 15)) * 1536 + ((lane >> 4) << 3);
  v8f C = {};
  for (int ks = 0; ks < 48; ++ks) {
    Frag A, B;
    A.h[0] = *(const v8h*)(ap + ks * 32);
    A.h[1] = *(const v8h*)(ap + ks * 32 + 16);
    B.h[0] = *(const v8h*)(bp + ks * 32);
    B.h[1] = *(const v8h*)(bp + ks * 32 + 16);
    C = wmma_f16(A.f16, B.f16, C);
  }
  const int n = nt * 16 + (lane & 15);
  const float bias = b1[n];
  const int rbase = rt * 16 + ((lane >> 4) << 3);
  #pragma unroll
  for (int v = 0; v < 8; ++v) {
    float hv = fmaxf(C[v] + bias, 0.f);
    hh[(size_t)(rbase + v) * 768 + n] = (_Float16)hv;
  }
}

// ---------------- kernel 6: out = (h @ W2^T + b2) or passthrough (WMMA, K=768) ----------------
__global__ __launch_bounds__(32)
void mlp2_kernel(const _Float16* __restrict__ hh, const _Float16* __restrict__ w2h,
                 const float* __restrict__ b2, const float* __restrict__ content,
                 const int* __restrict__ hasMatch, float* __restrict__ out)
{
  const int lane = threadIdx.x;
  const int nt = blockIdx.x, rt = blockIdx.y;
  const _Float16* ap = hh  + (size_t)(rt * 16 + (lane & 15)) * 768 + ((lane >> 4) << 3);
  const _Float16* bp = w2h + (size_t)(nt * 16 + (lane & 15)) * 768 + ((lane >> 4) << 3);
  v8f C = {};
  for (int ks = 0; ks < 24; ++ks) {
    Frag A, B;
    A.h[0] = *(const v8h*)(ap + ks * 32);
    A.h[1] = *(const v8h*)(ap + ks * 32 + 16);
    B.h[0] = *(const v8h*)(bp + ks * 32);
    B.h[1] = *(const v8h*)(bp + ks * 32 + 16);
    C = wmma_f16(A.f16, B.f16, C);
  }
  const int n = nt * 16 + (lane & 15);
  const float bias = b2[n];
  const int rbase = rt * 16 + ((lane >> 4) << 3);
  #pragma unroll
  for (int v = 0; v < 8; ++v) {
    const int row = rbase + v;
    const float val = C[v] + bias;
    out[(size_t)row * 768 + n] = hasMatch[row] ? val : content[(size_t)row * 768 + n];
  }
}

// ---------------- launcher ----------------
extern "C" void kernel_launch(void* const* d_in, const int* in_sizes, int n_in,
                              void* d_out, int out_size, void* d_ws, size_t ws_size,
                              hipStream_t stream) {
  const float* content = (const float*)d_in[0];   // (256, 768)
  const int*   tgt     = (const int*)  d_in[1];   // (256,)
  const float* train   = (const float*)d_in[2];   // (N, 768)
  const int*   spk     = (const int*)  d_in[3];   // (N,)
  const float* W1      = (const float*)d_in[4];   // (768, 1536)
  const float* b1      = (const float*)d_in[5];   // (768,)
  const float* W2      = (const float*)d_in[6];   // (768, 768)
  const float* b2      = (const float*)d_in[7];   // (768,)
  float* out = (float*)d_out;

  const int N = in_sizes[3];
  const int nChunk = (N + 63) / 64;

  // workspace carve-out (256B aligned)
  size_t off = 0;
  auto carve = [&](size_t bytes) { size_t o = off; off += (bytes + 255) & ~(size_t)255; return o; };
  char* ws = (char*)d_ws;
  _Float16* qnh  = (_Float16*)(ws + carve((size_t)256 * 768 * 2));
  _Float16* w1h  = (_Float16*)(ws + carve((size_t)768 * 1536 * 2));
  _Float16* w2h  = (_Float16*)(ws + carve((size_t)768 * 768 * 2));
  _Float16* comb = (_Float16*)(ws + carve((size_t)256 * 1536 * 2));
  _Float16* hh   = (_Float16*)(ws + carve((size_t)256 * 768 * 2));
  int*      hm   = (int*)     (ws + carve((size_t)256 * 4));
  float*    cV   = (float*)   (ws + carve((size_t)256 * nChunk * 5 * 4));
  int*      cI   = (int*)     (ws + carve((size_t)256 * nChunk * 5 * 4));

  qnorm_kernel<<<256, 256, 0, stream>>>(content, qnh);
  cvt_f16_kernel<<<(768 * 1536 + 255) / 256, 256, 0, stream>>>(W1, w1h, 768 * 1536);
  cvt_f16_kernel<<<(768 * 768  + 255) / 256, 256, 0, stream>>>(W2, w2h, 768 * 768);

  const size_t smem = 98304 + 768;  // B-frags + spk/inv/sum
  sim_topk_kernel<<<nChunk, 512, smem, stream>>>(qnh, train, spk, tgt, cV, cI, N, nChunk);

  merge_gather_kernel<<<256, 256, 0, stream>>>(cV, cI, train, content, comb, hm, nChunk);

  mlp1_kernel<<<dim3(48, 16), 32, 0, stream>>>(comb, w1h, b1, hh);
  mlp2_kernel<<<dim3(48, 16), 32, 0, stream>>>(hh, w2h, b2, content, hm, out);
}
